// MultiHeadSelfAttention_61692910239863
// MI455X (gfx1250) — matrix operand order repaired and checked by reading, not compiled, not run on hardware
//
#include <hip/hip_runtime.h>
#include <hip/hip_bf16.h>

#define DIM 768
#define NHEAD 12
#define HDIM 64
#define BATCH 16
#define SEQ 784
#define M_TOTAL (BATCH * SEQ)   /* 12544 */
#define QKV_N (3 * DIM)         /* 2304  */

typedef __bf16 bf16;
typedef __attribute__((ext_vector_type(16))) __bf16 v16bf;
typedef __attribute__((ext_vector_type(8)))  float  v8f;
typedef __attribute__((ext_vector_type(8)))  __bf16 bf16x8;

__device__ __forceinline__ v8f wmma_bf16(v16bf a, v16bf b, v8f c) {
  // 8 args: (neg_a, A, neg_b, B, c_mod, C, reuse_a, reuse_b)
  return __builtin_amdgcn_wmma_f32_16x16x32_bf16(false, a, false, b, (short)0, c,
                                                 false, false);
}

// CDNA5 async memory->LDS copy (16B per lane), tracked by ASYNCcnt.
// VDST = LDS byte address (VGPR), VADDR = 64-bit global address, saddr = off.
__device__ __forceinline__ void async_load_b128(void* lds_dst,
                                                const void* gsrc) {
  unsigned laddr =
      (unsigned)(size_t)(__attribute__((address_space(3))) void*)lds_dst;
  asm volatile("global_load_async_to_lds_b128 %0, %1, off" ::"v"(laddr),
               "v"(gsrc)
               : "memory");
}
__device__ __forceinline__ void wait_async() {
  asm volatile("s_wait_asynccnt 0x0" ::: "memory");
}

// ---------------------------------------------------------------------------
// elementwise f32 -> bf16
// ---------------------------------------------------------------------------
__global__ __launch_bounds__(256) void f32_to_bf16_kernel(
    const float* __restrict__ in, bf16* __restrict__ out, int n) {
  int i = blockIdx.x * blockDim.x + threadIdx.x;
  int stride = gridDim.x * blockDim.x;
  for (; i < n; i += stride) out[i] = (bf16)in[i];
}

// f32 [R][C] -> bf16 transposed [C][R] (coalesced reads, scattered b16 writes)
__global__ __launch_bounds__(256) void f32_to_bf16_T_kernel(
    const float* __restrict__ in, bf16* __restrict__ out, int R, int C) {
  int i = blockIdx.x * blockDim.x + threadIdx.x;
  int n = R * C;
  int stride = gridDim.x * blockDim.x;
  for (; i < n; i += stride) {
    int r = i / C, c = i - r * C;
    out[(size_t)c * R + r] = (bf16)in[i];
  }
}

// ---------------------------------------------------------------------------
// Tiled bf16 GEMM, B pre-transposed: C[M,N] = A[M,K] * Bt[N,K]^T + bias
// M%128==0, N%128==0, K%64==0. 256 thr = 8 waves; block tile 128x128, BK=64;
// wave -> 32 rows x 64 cols = 16 wmma per K-step. Staging via async-to-LDS.
// ---------------------------------------------------------------------------
template <bool STORE_BF16>
__global__ __launch_bounds__(256) void gemm_bf16_tn_kernel(
    const bf16* __restrict__ A, const bf16* __restrict__ Bt,
    const float* __restrict__ bias, float* __restrict__ Cf,
    bf16* __restrict__ Cb, int M, int N, int K) {
  __shared__ bf16 Al[128][72];    // 128 rows x 64 k (+8 pad, 16B-aligned rows)
  __shared__ bf16 Blt[128][72];   // 128 cols x 64 k (+8 pad)

  const int tid  = threadIdx.x;
  const int lane = tid & 31;
  const int wave = tid >> 5;
  const int wrow = wave & 3;      // 32-row group within 128
  const int wcol = wave >> 2;     // 64-col group within 128
  const int m0   = blockIdx.y * 128;
  const int n0   = blockIdx.x * 128;
  const int col  = lane & 15;
  const int kg   = lane >> 4;

  v8f acc[2][4];
#pragma unroll
  for (int r = 0; r < 2; ++r)
#pragma unroll
    for (int t = 0; t < 4; ++t)
#pragma unroll
      for (int i = 0; i < 8; ++i) acc[r][t][i] = 0.0f;

  for (int kb = 0; kb < K; kb += 64) {
    __syncthreads();
    // stage A (128x64) and Bt (128x64): 4 async b128 vectors per thread each
#pragma unroll
    for (int j = 0; j < 4; ++j) {
      int v = tid + j * 256;
      int r = v >> 3, c = (v & 7) * 8;
      async_load_b128(&Al[r][c], &A[(size_t)(m0 + r) * K + kb + c]);
      async_load_b128(&Blt[r][c], &Bt[(size_t)(n0 + r) * K + kb + c]);
    }
    if (kb + 64 < K) {  // global_prefetch_b8 of next tiles
      __builtin_prefetch(&A[(size_t)(m0 + (tid >> 1)) * K + kb + 64], 0, 0);
      __builtin_prefetch(&Bt[(size_t)(n0 + (tid >> 1)) * K + kb + 64], 0, 0);
    }
    wait_async();
    __syncthreads();

#pragma unroll
    for (int kc = 0; kc < 64; kc += 32) {
      // A fragments for the wave's two 16-row subtiles
      v16bf a[2];
#pragma unroll
      for (int r = 0; r < 2; ++r) {
        int row = wrow * 32 + r * 16 + col;
        bf16x8 lo = *(const bf16x8*)&Al[row][kc + 8 * kg];
        bf16x8 hi = *(const bf16x8*)&Al[row][kc + 16 + 8 * kg];
#pragma unroll
        for (int i = 0; i < 8; ++i) { a[r][i] = lo[i]; a[r][8 + i] = hi[i]; }
      }
#pragma unroll
      for (int t = 0; t < 4; ++t) {
        // B fragment: b[i] = Blt[c0][kc+8kg+i], b[8+i] = Blt[c0][kc+16+8kg+i]
        int c0 = wcol * 64 + t * 16 + col;
        bf16x8 lo = *(const bf16x8*)&Blt[c0][kc + 8 * kg];
        bf16x8 hi = *(const bf16x8*)&Blt[c0][kc + 16 + 8 * kg];
        v16bf b;
#pragma unroll
        for (int i = 0; i < 8; ++i) { b[i] = lo[i]; b[8 + i] = hi[i]; }
        acc[0][t] = wmma_bf16(a[0], b, acc[0][t]);
        acc[1][t] = wmma_bf16(a[1], b, acc[1][t]);
      }
    }
  }

  // epilogue: C row = i + 8*kg, col = lane&15
#pragma unroll
  for (int r = 0; r < 2; ++r)
#pragma unroll
    for (int t = 0; t < 4; ++t) {
      int cg = n0 + wcol * 64 + t * 16 + col;
      float bv = bias[cg];
#pragma unroll
      for (int i = 0; i < 8; ++i) {
        int rg = m0 + wrow * 32 + r * 16 + i + 8 * kg;
        float v = acc[r][t][i] + bv;
        if (STORE_BF16)
          Cb[(size_t)rg * N + cg] = (bf16)v;
        else
          Cf[(size_t)rg * N + cg] = v;
      }
    }
}

// ---------------------------------------------------------------------------
// Flash attention: block = (128-row tile, b*NHEAD+h); 8 waves, wave = 16 rows.
// 64-key blocks: 8 score wmma + 8 PV wmma per block. K staged via async-to-LDS
// (overlapped with V transpose staging); out-of-range rows address-clamped,
// invalid keys masked in softmax.
// QKV layout: qkv[b*SEQ+n][sel*DIM + h*HDIM + d]. Out: Ob[b*SEQ+n][h*64+d].
// ---------------------------------------------------------------------------
__global__ __launch_bounds__(256) void attention_kernel(
    const bf16* __restrict__ qkv, bf16* __restrict__ Ob) {
  __shared__ bf16 Kl[64][72];      // [key][d]
  __shared__ bf16 Vlt[64][72];     // [d][key]  (transposed)
  __shared__ bf16 Pl[8][16][72];   // per-wave P scratch: [row][key]

  const int tid  = threadIdx.x;
  const int lane = tid & 31;
  const int wave = tid >> 5;
  const int col  = lane & 15;
  const int kg   = lane >> 4;

  const int bh = blockIdx.y;
  const int b  = bh / NHEAD;
  const int h  = bh % NHEAD;
  const int rowBase = blockIdx.x * 128 + wave * 16;

  const size_t rs = QKV_N;
  const bf16* qbase = qkv + (size_t)b * SEQ * rs + (size_t)h * HDIM;

  // Q A-fragments (d=[0,32) and [32,64)), rows = rowBase + (lane&15), clamped
  v16bf q0, q1;
  {
    int r0 = rowBase + col;
    int r = (r0 < SEQ) ? r0 : (SEQ - 1);
    const bf16* qr = qbase + (size_t)r * rs;
    bf16x8 a0 = *(const bf16x8*)&qr[8 * kg];
    bf16x8 a1 = *(const bf16x8*)&qr[16 + 8 * kg];
    bf16x8 a2 = *(const bf16x8*)&qr[32 + 8 * kg];
    bf16x8 a3 = *(const bf16x8*)&qr[48 + 8 * kg];
#pragma unroll
    for (int i = 0; i < 8; ++i) {
      q0[i] = a0[i]; q0[8 + i] = a1[i];
      q1[i] = a2[i]; q1[8 + i] = a3[i];
    }
  }

  float m[8], l[8];
  v8f o[4];
#pragma unroll
  for (int i = 0; i < 8; ++i) { m[i] = -1e30f; l[i] = 0.0f; }
#pragma unroll
  for (int t = 0; t < 4; ++t)
#pragma unroll
    for (int i = 0; i < 8; ++i) o[t][i] = 0.0f;

  const float scale = 0.125f;  // 1/sqrt(64)

  for (int jb = 0; jb < SEQ; jb += 64) {  // 13 blocks (784 = 12*64 + 16)
    __syncthreads();
    // issue async K staging first, then do V transpose staging to overlap
#pragma unroll
    for (int j = 0; j < 2; ++j) {
      int v = tid + j * 256;
      int r = v >> 3;          // key 0..63
      int c = (v & 7) * 8;     // d
      int k0 = jb + r;
      int krow = (k0 < SEQ) ? k0 : (SEQ - 1);
      async_load_b128(&Kl[r][c], &qbase[DIM + (size_t)krow * rs + c]);
    }
#pragma unroll
    for (int j = 0; j < 2; ++j) {
      int v = tid + j * 256;
      int r = v >> 3;
      int c = (v & 7) * 8;
      int k0 = jb + r;
      int krow = (k0 < SEQ) ? k0 : (SEQ - 1);
      bf16x8 vz = *(const bf16x8*)&qbase[2 * DIM + (size_t)krow * rs + c];
#pragma unroll
      for (int e = 0; e < 8; ++e) Vlt[c + e][r] = vz[e];
    }
    wait_async();
    __syncthreads();

    // scores: four 16-key subtiles, 2 chained wmma each (K dim = d = 64)
    v8f s[4];
#pragma unroll
    for (int si = 0; si < 4; ++si)
#pragma unroll
      for (int i = 0; i < 8; ++i) s[si][i] = 0.0f;
#pragma unroll
    for (int si = 0; si < 4; ++si) {
      const bf16* kr = &Kl[si * 16 + col][0];  // B[d][key] = Kl[key][d]
      bf16x8 l0 = *(const bf16x8*)&kr[8 * kg];
      bf16x8 l1 = *(const bf16x8*)&kr[16 + 8 * kg];
      bf16x8 l2 = *(const bf16x8*)&kr[32 + 8 * kg];
      bf16x8 l3 = *(const bf16x8*)&kr[48 + 8 * kg];
      v16bf bk0, bk1;
#pragma unroll
      for (int i = 0; i < 8; ++i) {
        bk0[i] = l0[i]; bk0[8 + i] = l1[i];
        bk1[i] = l2[i]; bk1[8 + i] = l3[i];
      }
      s[si] = wmma_bf16(q0, bk0, s[si]);
      s[si] = wmma_bf16(q1, bk1, s[si]);
    }

    bool okv[4];
#pragma unroll
    for (int si = 0; si < 4; ++si) okv[si] = (jb + si * 16 + col) < SEQ;

    // online softmax; C-layout row = i + 8*kg, 16-lane row groups
#pragma unroll
    for (int i = 0; i < 8; ++i) {
      float sv[4];
#pragma unroll
      for (int si = 0; si < 4; ++si)
        sv[si] = okv[si] ? s[si][i] * scale : -1e30f;
      float mb = fmaxf(fmaxf(sv[0], sv[1]), fmaxf(sv[2], sv[3]));
#pragma unroll
      for (int off = 1; off < 16; off <<= 1)
        mb = fmaxf(mb, __shfl_xor(mb, off, 32));
      float mn = fmaxf(m[i], mb);
      float alpha = __expf(m[i] - mn);
      float p[4], ls = 0.0f;
#pragma unroll
      for (int si = 0; si < 4; ++si) { p[si] = __expf(sv[si] - mn); ls += p[si]; }
#pragma unroll
      for (int off = 1; off < 16; off <<= 1) ls += __shfl_xor(ls, off, 32);
      l[i] = l[i] * alpha + ls;
      m[i] = mn;
#pragma unroll
      for (int t = 0; t < 4; ++t) o[t][i] *= alpha;
#pragma unroll
      for (int si = 0; si < 4; ++si)
        Pl[wave][i + 8 * kg][si * 16 + col] = (bf16)p[si];
    }
    __syncthreads();

    // P A-fragments (16 x 64 keys -> 2 frags), contiguous b128 reads
    v16bf pa0, pa1;
    {
      const bf16* pr = &Pl[wave][col][0];
      bf16x8 a0 = *(const bf16x8*)&pr[8 * kg];
      bf16x8 a1 = *(const bf16x8*)&pr[16 + 8 * kg];
      bf16x8 a2 = *(const bf16x8*)&pr[32 + 8 * kg];
      bf16x8 a3 = *(const bf16x8*)&pr[48 + 8 * kg];
#pragma unroll
      for (int i = 0; i < 8; ++i) {
        pa0[i] = a0[i]; pa0[8 + i] = a1[i];
        pa1[i] = a2[i]; pa1[8 + i] = a3[i];
      }
    }
    // O += P @ V : B[key][d] fragments from Vlt[d][key] (contiguous)
#pragma unroll
    for (int t = 0; t < 4; ++t) {
      const bf16* vr = &Vlt[t * 16 + col][0];
      bf16x8 c0 = *(const bf16x8*)&vr[8 * kg];
      bf16x8 c1 = *(const bf16x8*)&vr[16 + 8 * kg];
      bf16x8 c2 = *(const bf16x8*)&vr[32 + 8 * kg];
      bf16x8 c3 = *(const bf16x8*)&vr[48 + 8 * kg];
      v16bf bv0, bv1;
#pragma unroll
      for (int i = 0; i < 8; ++i) {
        bv0[i] = c0[i]; bv0[8 + i] = c1[i];
        bv1[i] = c2[i]; bv1[8 + i] = c3[i];
      }
      o[t] = wmma_bf16(pa0, bv0, o[t]);
      o[t] = wmma_bf16(pa1, bv1, o[t]);
    }
  }

  // normalize + store (bnd layout, bf16 feed for output projection)
#pragma unroll
  for (int i = 0; i < 8; ++i) {
    int n = rowBase + i + 8 * kg;
    if (n >= SEQ) continue;
    float rinv = (l[i] > 0.0f) ? 1.0f / l[i] : 0.0f;
#pragma unroll
    for (int t = 0; t < 4; ++t) {
      Ob[(size_t)(b * SEQ + n) * DIM + h * HDIM + t * 16 + col] =
          (bf16)(o[t][i] * rinv);
    }
  }
}

// ---------------------------------------------------------------------------
extern "C" void kernel_launch(void* const* d_in, const int* in_sizes, int n_in,
                              void* d_out, int out_size, void* d_ws,
                              size_t ws_size, hipStream_t stream) {
  const float* x    = (const float*)d_in[0];
  const float* Wqkv = (const float*)d_in[1];
  const float* bqkv = (const float*)d_in[2];
  const float* Wo   = (const float*)d_in[3];
  const float* bo   = (const float*)d_in[4];
  float* out = (float*)d_out;

  // workspace layout (bytes, 16B-aligned offsets)
  char* ws = (char*)d_ws;
  bf16* Xb     = (bf16*)(ws);                 // 12544x768  bf16 : 19267584 B
  bf16* Wqkvt  = (bf16*)(ws + 19267584);      // 2304x768   bf16 :  3538944 B
  bf16* Wot    = (bf16*)(ws + 22806528);      // 768x768    bf16 :  1179648 B
  bf16* QKVb   = (bf16*)(ws + 23986176);      // 12544x2304 bf16 : 57802752 B
  bf16* Oa     = (bf16*)(ws + 81788928);      // 12544x768  bf16 : 19267584 B

  f32_to_bf16_kernel<<<2048, 256, 0, stream>>>(x, Xb, M_TOTAL * DIM);
  f32_to_bf16_T_kernel<<<1024, 256, 0, stream>>>(Wqkv, Wqkvt, DIM, QKV_N);
  f32_to_bf16_T_kernel<<<256, 256, 0, stream>>>(Wo, Wot, DIM, DIM);

  // QKV projection: [12544,768] x [768,2304] + bqkv -> bf16
  gemm_bf16_tn_kernel<true>
      <<<dim3(QKV_N / 128, M_TOTAL / 128), 256, 0, stream>>>(
          Xb, Wqkvt, bqkv, nullptr, QKVb, M_TOTAL, QKV_N, DIM);

  // flash attention per (b,h); 128-row tiles (784 -> 7 tiles)
  attention_kernel<<<dim3((SEQ + 127) / 128, BATCH * NHEAD), 256, 0, stream>>>(
      QKVb, Oa);

  // output projection: [12544,768] x [768,768] + bo -> f32
  gemm_bf16_tn_kernel<false>
      <<<dim3(DIM / 128, M_TOTAL / 128), 256, 0, stream>>>(
          Oa, Wot, bo, out, nullptr, M_TOTAL, DIM, DIM);
}
